// AdaptiveModalityEncoder_21337397527133
// MI455X (gfx1250) — matrix operand order repaired and checked by reading, not compiled, not run on hardware
//
#include <hip/hip_runtime.h>

typedef __attribute__((ext_vector_type(16))) __bf16   v16bf;
typedef __attribute__((ext_vector_type(8)))  __bf16   v8bf;
typedef __attribute__((ext_vector_type(2)))  __bf16   bf16x2;
typedef __attribute__((ext_vector_type(8)))  float    v8f;
typedef __attribute__((ext_vector_type(4)))  unsigned u32x4;
typedef __attribute__((ext_vector_type(4)))  int      v4i;

#if defined(__gfx1250__) && __has_builtin(__builtin_amdgcn_global_load_async_to_lds_b128)
#define USE_ASYNC_LDS 1
#else
#define USE_ASYNC_LDS 0
#endif

// ---------- helpers ----------------------------------------------------------

// Split a pair of fp32 into packed bf16 hi and bf16 lo (residual) dwords.
__device__ __forceinline__ void split_pack(float x, float y, unsigned &hi, unsigned &lo) {
  __bf16 xh = (__bf16)x, yh = (__bf16)y;
  __bf16 xl = (__bf16)(x - (float)xh);
  __bf16 yl = (__bf16)(y - (float)yh);
  bf16x2 h, l;
  h[0] = xh; h[1] = yh;
  l[0] = xl; l[1] = yl;
  hi = __builtin_bit_cast(unsigned, h);
  lo = __builtin_bit_cast(unsigned, l);
}

// Fragment tile layout (per 16x32 A-tile / 32x16 B-tile, 256 u32):
//   [half(2)][lane(32)][4 u32] -> two conflict-free ds_load_b128 per lane.
__device__ __forceinline__ v16bf frag_from_lds(const unsigned* lds, int base, int lane) {
  u32x4 c0 = *(const u32x4*)(lds + base + lane * 4);
  u32x4 c1 = *(const u32x4*)(lds + base + 128 + lane * 4);
  v8bf a = __builtin_bit_cast(v8bf, c0);
  v8bf b = __builtin_bit_cast(v8bf, c1);
  return __builtin_shufflevector(a, b, 0,1,2,3,4,5,6,7,8,9,10,11,12,13,14,15);
}

__device__ __forceinline__ float gelu_tanh(float x) {
  float x3 = x * x * x;
  return 0.5f * x * (1.0f + tanhf(0.7978845608028654f * (x + 0.044715f * x3)));
}

// ---------- pre-conversion: fp32 -> bf16 hi/lo, WMMA fragment-tile order -----

// A-operand (row-major [R,C]) -> tiles [kstep=C/32][mtile=R/16][256 u32]
// lane = m%16 + 16*bit3(k); elem idx = k%8 + 8*bit4(k); half = bit4(k)
__global__ __launch_bounds__(256) void convert_a_frag(
    const float* __restrict__ X, unsigned* __restrict__ Ohi, unsigned* __restrict__ Olo,
    int C, int lgGpr /* log2(C/8) */, int Rtiles) {
  int g   = blockIdx.x * 256 + threadIdx.x;  // one 8-element group (same row, k..k+7)
  int row = g >> lgGpr;
  int k   = (g & ((1 << lgGpr) - 1)) << 3;
  const float* p = X + (size_t)row * C + k;
  float4 a = *(const float4*)p;
  float4 b = *(const float4*)(p + 4);
  unsigned h0,l0,h1,l1,h2,l2,h3,l3;
  split_pack(a.x, a.y, h0, l0); split_pack(a.z, a.w, h1, l1);
  split_pack(b.x, b.y, h2, l2); split_pack(b.z, b.w, h3, l3);
  int lane_a = (row & 15) | (((k >> 3) & 1) << 4);
  size_t base = ((size_t)(k >> 5) * Rtiles + (row >> 4)) * 256
              + (((k >> 4) & 1) << 7) + lane_a * 4;
  u32x4 hv = {h0, h1, h2, h3}, lv = {l0, l1, l2, l3};
  *(u32x4*)(Ohi + base) = hv;
  *(u32x4*)(Olo + base) = lv;
}

// B-operand (row-major [K,N]) -> tiles [kstep=K/32][ntile=N/16][256 u32]
// lane = n%16 + 16*bit3(k); elem idx = k%8 + 8*bit4(k); half = bit4(k)
__global__ __launch_bounds__(256) void convert_b_frag(
    const float* __restrict__ W, unsigned* __restrict__ Ohi, unsigned* __restrict__ Olo,
    int N, int lgN) {
  int g = blockIdx.x * 256 + threadIdx.x;    // one 8-element group (same col, k..k+7)
  int n = g & (N - 1);
  int k = (g >> lgN) << 3;
  const float* p = W + (size_t)k * N + n;
  float w0 = p[0],             w1 = p[(size_t)N],
        w2 = p[(size_t)2 * N], w3 = p[(size_t)3 * N],
        w4 = p[(size_t)4 * N], w5 = p[(size_t)5 * N],
        w6 = p[(size_t)6 * N], w7 = p[(size_t)7 * N];
  unsigned h0,l0,h1,l1,h2,l2,h3,l3;
  split_pack(w0, w1, h0, l0); split_pack(w2, w3, h1, l1);
  split_pack(w4, w5, h2, l2); split_pack(w6, w7, h3, l3);
  int lane_b = (n & 15) | (((k >> 3) & 1) << 4);
  size_t base = ((size_t)(k >> 5) * (N >> 4) + (n >> 4)) * 256
              + (((k >> 4) & 1) << 7) + lane_b * 4;
  u32x4 hv = {h0, h1, h2, h3}, lv = {l0, l1, l2, l3};
  *(u32x4*)(Ohi + base) = hv;
  *(u32x4*)(Olo + base) = lv;
}

// ---------- main WMMA GEMM ---------------------------------------------------
// C = A @ B (+bias) from pre-swizzled bf16x2 fragment tiles.
// OUT_FRAG: gelu epilogue, output written as bf16x2 fragment tiles (layer 1).
// else:     row-mask epilogue, fp32 row-major output (layer 2).
template <bool OUT_FRAG>
__global__ __launch_bounds__(256) void wmma_gemm_frag(
    const unsigned* __restrict__ Ahi, const unsigned* __restrict__ Alo,
    const unsigned* __restrict__ Bhi, const unsigned* __restrict__ Blo,
    const float* __restrict__ bias, int Kdim, int Mtiles, int Ntiles,
    float* __restrict__ Cf, unsigned* __restrict__ Ohi, unsigned* __restrict__ Olo,
    const float* __restrict__ selmask, const int* __restrict__ midx_ptr)
{
  // 64KB: 2 stages x {Ahi,Alo,Bhi,Blo} x 2048 u32; reused as 128x128 fp32 tile.
  __shared__ unsigned lds[16384];

  const int tid  = threadIdx.x;
  const int lane = tid & 31;
  const int wid  = tid >> 5;
  const int wm   = wid & 3;    // 4 wave-rows of 32
  const int wn   = wid >> 2;   // 2 wave-cols of 64
  const int row0 = blockIdx.y * 128, col0 = blockIdx.x * 128;
  const int mt0  = row0 >> 4,  nt0  = col0 >> 4;

#if !USE_ASYNC_LDS
  u32x4 pf[8];  // register prefetch staging (fallback path only)
#endif

  auto issueLoads = [&](int stage, int s) {
    const size_t ab = ((size_t)s * Mtiles + mt0) * 256 + tid * 8;
    const size_t bb = ((size_t)s * Ntiles + nt0) * 256 + tid * 8;
    const unsigned* src[4] = {Ahi + ab, Alo + ab, Bhi + bb, Blo + bb};
#if USE_ASYNC_LDS
    typedef __attribute__((address_space(1))) v4i* gv4p;
    typedef __attribute__((address_space(3))) unsigned* l32p;
    typedef __attribute__((address_space(3))) v4i* lv4p;
    unsigned* dst = lds + stage * 8192 + tid * 8;
    #pragma unroll
    for (int r = 0; r < 4; ++r) {
      gv4p sg = (gv4p)(size_t)(const void*)src[r];
      lv4p dl = (lv4p)(l32p)(dst + r * 2048);
      __builtin_amdgcn_global_load_async_to_lds_b128(sg,     dl,     0, 0);
      __builtin_amdgcn_global_load_async_to_lds_b128(sg + 1, dl + 1, 0, 0);
    }
#else
    #pragma unroll
    for (int r = 0; r < 4; ++r) {
      pf[2 * r]     = *(const u32x4*)(src[r]);
      pf[2 * r + 1] = *(const u32x4*)(src[r] + 4);
    }
    (void)stage;
#endif
  };

  auto commitLoads = [&](int stage) {
#if !USE_ASYNC_LDS
    unsigned* dst = lds + stage * 8192 + tid * 8;
    #pragma unroll
    for (int r = 0; r < 4; ++r) {
      *(u32x4*)(dst + r * 2048)     = pf[2 * r];
      *(u32x4*)(dst + r * 2048 + 4) = pf[2 * r + 1];
    }
#else
    (void)stage;
#endif
  };

  auto waitAsync = [&]() {
#if USE_ASYNC_LDS
 #if __has_builtin(__builtin_amdgcn_s_wait_asynccnt)
    __builtin_amdgcn_s_wait_asynccnt(0);
 #else
    asm volatile("s_wait_asynccnt 0x0" ::: "memory");
 #endif
#endif
  };

  v8f acc[2][4];
  #pragma unroll
  for (int i = 0; i < 2; ++i)
    #pragma unroll
    for (int j = 0; j < 4; ++j)
      acc[i][j] = v8f{};

  auto compute = [&](int stage) {
    const unsigned* LAh = lds + stage * 8192;
    const unsigned* LAl = LAh + 2048;
    const unsigned* LBh = LAh + 4096;
    const unsigned* LBl = LAh + 6144;
    v16bf ah[2], al[2], bh[4], bl[4];
    #pragma unroll
    for (int mi = 0; mi < 2; ++mi) {
      int mt = wm * 2 + mi;
      ah[mi] = frag_from_lds(LAh, mt << 8, lane);
      al[mi] = frag_from_lds(LAl, mt << 8, lane);
    }
    #pragma unroll
    for (int ni = 0; ni < 4; ++ni) {
      int nt = wn * 4 + ni;
      bh[ni] = frag_from_lds(LBh, nt << 8, lane);
      bl[ni] = frag_from_lds(LBl, nt << 8, lane);
    }
    #pragma unroll
    for (int mi = 0; mi < 2; ++mi)
      #pragma unroll
      for (int ni = 0; ni < 4; ++ni) {
        acc[mi][ni] = __builtin_amdgcn_wmma_f32_16x16x32_bf16(
            false, ah[mi], false, bh[ni], (short)0, acc[mi][ni], false, false);
        acc[mi][ni] = __builtin_amdgcn_wmma_f32_16x16x32_bf16(
            false, ah[mi], false, bl[ni], (short)0, acc[mi][ni], false, false);
        acc[mi][ni] = __builtin_amdgcn_wmma_f32_16x16x32_bf16(
            false, al[mi], false, bh[ni], (short)0, acc[mi][ni], false, false);
      }
  };

  const int nsteps = Kdim >> 5;
  issueLoads(0, 0);
  commitLoads(0);
  waitAsync();
  __syncthreads();
  for (int s = 0; s < nsteps; ++s) {
    const int  cur  = s & 1;
    const bool more = (s + 1) < nsteps;
    if (more) issueLoads(cur ^ 1, s + 1);   // overlap with WMMA stream
    compute(cur);
    if (more) commitLoads(cur ^ 1);
    waitAsync();
    __syncthreads();
  }

  const int ln = lane & 15, lh = lane >> 4;
  if constexpr (OUT_FRAG) {
    // Stage bias+GELU result as fp32 128x128 tile in LDS, then repack to
    // bf16 hi/lo A-fragment tiles in global memory (coalesced b128 stores).
    float* tile = (float*)lds;
    #pragma unroll
    for (int ni = 0; ni < 4; ++ni) {
      int lcol = wn * 64 + ni * 16 + ln;
      float bv = bias[col0 + lcol];
      #pragma unroll
      for (int mi = 0; mi < 2; ++mi)
        #pragma unroll
        for (int r = 0; r < 8; ++r) {
          int lrow = wm * 32 + mi * 16 + r + (lh << 3);
          tile[lrow * 128 + lcol] = gelu_tanh(acc[mi][ni][r] + bv);
        }
    }
    __syncthreads();
    #pragma unroll
    for (int p = 0; p < 8; ++p) {
      int g      = tid + (p << 8);          // 2048 groups of 8 elements
      int tileid = g >> 6;                  // [ks_l(4)][mt_l(8)]
      int ks_l   = tileid >> 3, mt_l = tileid & 7;
      int idx    = g & 63;
      int half   = idx >> 5,    la   = idx & 31;
      int row_l  = mt_l * 16 + (la & 15);
      int col_l  = ks_l * 32 + half * 16 + ((la >> 4) << 3);
      const float* sp = tile + row_l * 128 + col_l;
      float4 a = *(const float4*)sp;
      float4 b = *(const float4*)(sp + 4);
      unsigned h0,l0,h1,l1,h2,l2,h3,l3;
      split_pack(a.x, a.y, h0, l0); split_pack(a.z, a.w, h1, l1);
      split_pack(b.x, b.y, h2, l2); split_pack(b.z, b.w, h3, l3);
      size_t ob = ((size_t)((col0 >> 5) + ks_l) * Mtiles + (mt0 + mt_l)) * 256
                + (half << 7) + la * 4;
      u32x4 hv = {h0, h1, h2, h3}, lv = {l0, l1, l2, l3};
      *(u32x4*)(Ohi + ob) = hv;
      *(u32x4*)(Olo + ob) = lv;
    }
  } else {
    const int Nn  = Ntiles << 4;
    const int mdx = midx_ptr[0];
    #pragma unroll
    for (int ni = 0; ni < 4; ++ni) {
      int gcol = col0 + wn * 64 + ni * 16 + ln;
      float bv = bias[gcol];
      #pragma unroll
      for (int mi = 0; mi < 2; ++mi)
        #pragma unroll
        for (int r = 0; r < 8; ++r) {
          int grow = row0 + wm * 32 + mi * 16 + r + (lh << 3);
          float v = acc[mi][ni][r] + bv;
          float m = (selmask[(size_t)grow * 4 + mdx] > 0.5f) ? 1.0f : 0.0f;
          Cf[(size_t)grow * Nn + gcol] = v * m;
        }
    }
  }
}

// ---------- launch -----------------------------------------------------------

extern "C" void kernel_launch(void* const* d_in, const int* in_sizes, int n_in,
                              void* d_out, int out_size, void* d_ws, size_t ws_size,
                              hipStream_t stream) {
  (void)in_sizes; (void)n_in; (void)out_size; (void)ws_size;
  const float* X   = (const float*)d_in[0];  // [16384, 1024]
  const float* sel = (const float*)d_in[1];  // [16384, 4]
  const float* W1  = (const float*)d_in[2];  // [1024, 2048]
  const float* b1  = (const float*)d_in[3];  // [2048]
  const float* W2  = (const float*)d_in[4];  // [2048, 1024]
  const float* b2  = (const float*)d_in[5];  // [1024]
  const int*   mi  = (const int*)d_in[6];    // scalar modality_idx
  float* out = (float*)d_out;                // [16384, 1024]

  char* ws = (char*)d_ws;                    // 208 MB total
  unsigned* Xhi = (unsigned*)(ws);                          // 32 MB
  unsigned* Xlo = (unsigned*)(ws + (((size_t)32)  << 20));  // 32 MB
  unsigned* W1h = (unsigned*)(ws + (((size_t)64)  << 20));  //  4 MB
  unsigned* W1l = (unsigned*)(ws + (((size_t)68)  << 20));  //  4 MB
  unsigned* W2h = (unsigned*)(ws + (((size_t)72)  << 20));  //  4 MB
  unsigned* W2l = (unsigned*)(ws + (((size_t)76)  << 20));  //  4 MB
  unsigned* H1h = (unsigned*)(ws + (((size_t)80)  << 20));  // 64 MB
  unsigned* H1l = (unsigned*)(ws + (((size_t)144) << 20));  // 64 MB

  dim3 blk(256);
  // X:  R=16384, C=1024  -> 2M groups
  convert_a_frag<<<8192, blk, 0, stream>>>(X, Xhi, Xlo, 1024, 7, 1024);
  // W1: K=1024, N=2048 ; W2: K=2048, N=1024 -> 256K groups each
  convert_b_frag<<<1024, blk, 0, stream>>>(W1, W1h, W1l, 2048, 11);
  convert_b_frag<<<1024, blk, 0, stream>>>(W2, W2h, W2l, 1024, 10);

  // Layer 1: [16384,1024] @ [1024,2048], gelu, -> H1 fragment tiles
  wmma_gemm_frag<true><<<dim3(16, 128), blk, 0, stream>>>(
      Xhi, Xlo, W1h, W1l, b1, 1024, 1024, 128, nullptr, H1h, H1l, nullptr, nullptr);
  // Layer 2: [16384,2048] @ [2048,1024], row mask, -> fp32 out
  wmma_gemm_frag<false><<<dim3(8, 128), blk, 0, stream>>>(
      H1h, H1l, W2h, W2l, b2, 2048, 1024, 64, out, nullptr, nullptr, sel, mi);
}
